// MultiHeadAttention_31009663877401
// MI455X (gfx1250) — matrix operand order repaired and checked by reading, not compiled, not run on hardware
//
#include <hip/hip_runtime.h>

// ---------------------------------------------------------------------------
// MHA for MI455X (gfx1250): f16 WMMA (V_WMMA_F32_16X16X32_F16) everywhere,
// fused scores+softmax in ~259KB dynamic LDS, P written once / read once.
// All WMMA loops are software-pipelined (double-buffered fragments) so the
// compiler does not have to drain loadcnt to zero before each matrix op.
// ---------------------------------------------------------------------------

#define N_NODES 4096
#define HID     512
#define NHEAD   8
#define HDIM    64

typedef __attribute__((ext_vector_type(16))) _Float16 v16h;
typedef __attribute__((ext_vector_type(8)))  _Float16 v8h;
typedef __attribute__((ext_vector_type(8)))  float    v8f;

static __device__ __forceinline__ v8f vzero8() {
  v8f z;
#pragma unroll
  for (int i = 0; i < 8; ++i) z[i] = 0.0f;
  return z;
}

// D = A(16x32) * B(32x16) + C, f32 accum
static __device__ __forceinline__ v8f wmma_f16(v16h a, v16h b, v8f c) {
  return __builtin_amdgcn_wmma_f32_16x16x32_f16(false, a, false, b,
                                                (short)0, c, false, false);
}

// A fragment 16x32 f16, row-major source, tile origin p0, leading dim lda.
// Lane l: row = l&15, half h = l>>4; element i holds K = (i>>3)*16 + h*8 + (i&7).
static __device__ __forceinline__ v16h load_a16(const _Float16* __restrict__ p0, int lda) {
  const int lane = threadIdx.x & 31;
  const int r = lane & 15, hh = lane >> 4;
  const _Float16* p = p0 + (size_t)r * lda + hh * 8;
  v8h lo = *(const v8h*)(p);        // K = hh*8 .. hh*8+7
  v8h hi = *(const v8h*)(p + 16);   // K = 16+hh*8 ..
  v16h o;
#pragma unroll
  for (int i = 0; i < 8; ++i) { o[i] = lo[i]; o[i + 8] = hi[i]; }
  return o;
}

// A fragment from fp32 source (the softmaxed P matrix), converted to f16.
static __device__ __forceinline__ v16h load_a32(const float* __restrict__ p0, int lda) {
  const int lane = threadIdx.x & 31;
  const int r = lane & 15, hh = lane >> 4;
  const float* p = p0 + (size_t)r * lda + hh * 8;
  v8f lo = *(const v8f*)(p);
  v8f hi = *(const v8f*)(p + 16);
  v16h o;
#pragma unroll
  for (int i = 0; i < 8; ++i) { o[i] = (_Float16)lo[i]; o[i + 8] = (_Float16)hi[i]; }
  return o;
}

// B fragment 32x16 f16 from a TRANSPOSED (column-major) matrix:
// p0 points at BT[n0][k0], BT row-major [N][K], ldb = K-stride.
// Lane l: col = l&15, half h = l>>4; element i holds K = (i>>3)*16 + h*8 + (i&7).
static __device__ __forceinline__ v16h load_bt16(const _Float16* __restrict__ p0, int ldb) {
  const int lane = threadIdx.x & 31;
  const int c = lane & 15, hh = lane >> 4;
  const _Float16* p = p0 + (size_t)c * ldb + hh * 8;
  v8h lo = *(const v8h*)(p);
  v8h hi = *(const v8h*)(p + 16);
  v16h o;
#pragma unroll
  for (int i = 0; i < 8; ++i) { o[i] = lo[i]; o[i + 8] = hi[i]; }
  return o;
}

// ---------------------------------------------------------------------------
// Stage 0: casts
// ---------------------------------------------------------------------------
__global__ void cast_f32_to_f16(const float* __restrict__ in, _Float16* __restrict__ out, int n) {
  int i = blockIdx.x * blockDim.x + threadIdx.x;
  if (i < n) out[i] = (_Float16)in[i];
}

// W [K][Nc] fp32  ->  WT [Nc][K] f16  (column-major B operand)
__global__ void cast_transpose_f16(const float* __restrict__ W, _Float16* __restrict__ WT,
                                   int K, int Nc) {
  int i = blockIdx.x * blockDim.x + threadIdx.x;
  if (i < K * Nc) {
    int r = i / Nc, c = i - r * Nc;
    WT[(size_t)c * K + r] = (_Float16)W[i];
  }
}

// ---------------------------------------------------------------------------
// Stage 1: Q/K/V projections.  grid(32, 8, 3), block 256 (8 waves).
// Wave w: rows bx*128 + w*16 .. +15 ; cols by*64 .. +63 (4 WMMA col-tiles).
// z: 0=Q (scaled by 1/sqrt(d)), 1=K, 2=V (stored transposed [h][d][N]).
// Double-buffered k-loop: next fragments load while current WMMAs execute.
// ---------------------------------------------------------------------------
__global__ void qkv_gemm(const _Float16* __restrict__ x16,
                         const _Float16* __restrict__ WqT,
                         const _Float16* __restrict__ WkT,
                         const _Float16* __restrict__ WvT,
                         const float* __restrict__ bq,
                         const float* __restrict__ bk,
                         const float* __restrict__ bv,
                         _Float16* __restrict__ q16,
                         _Float16* __restrict__ k16,
                         _Float16* __restrict__ vT16) {
  const int z = blockIdx.z;
  const _Float16* WT = (z == 0) ? WqT : (z == 1) ? WkT : WvT;
  const float* bias  = (z == 0) ? bq  : (z == 1) ? bk  : bv;

  const int w    = threadIdx.x >> 5;
  const int lane = threadIdx.x & 31;
  const int hh   = lane >> 4, c16 = lane & 15;
  const int n0   = blockIdx.x * 128 + w * 16;
  const int c0   = blockIdx.y * 64;

  const _Float16* Ap = x16 + (size_t)n0 * HID;
  const _Float16* Bp = WT + (size_t)c0 * HID;

  v8f acc[4];
#pragma unroll
  for (int t = 0; t < 4; ++t) acc[t] = vzero8();

  v16h a = load_a16(Ap, HID);
  v16h b[4];
#pragma unroll
  for (int t = 0; t < 4; ++t) b[t] = load_bt16(Bp + (size_t)(t * 16) * HID, HID);

  for (int kk = 0; kk < HID / 32; ++kk) {
    const int kn = ((kk + 1) & (HID / 32 - 1)) * 32;  // wrap: always in-bounds
    v16h an = load_a16(Ap + kn, HID);
    v16h bn[4];
#pragma unroll
    for (int t = 0; t < 4; ++t) bn[t] = load_bt16(Bp + (size_t)(t * 16) * HID + kn, HID);
#pragma unroll
    for (int t = 0; t < 4; ++t) acc[t] = wmma_f16(a, b[t], acc[t]);
    a = an;
#pragma unroll
    for (int t = 0; t < 4; ++t) b[t] = bn[t];
  }

#pragma unroll
  for (int t = 0; t < 4; ++t) {
#pragma unroll
    for (int e = 0; e < 8; ++e) {
      const int row = n0 + e + 8 * hh;
      const int col = c0 + t * 16 + c16;
      float val = acc[t][e] + bias[col];
      if (z == 0) {
        val *= 0.125f;  // 1/sqrt(64) folded into Q
        q16[((size_t)(col >> 6) * N_NODES + row) * HDIM + (col & 63)] = (_Float16)val;
      } else if (z == 1) {
        k16[((size_t)(col >> 6) * N_NODES + row) * HDIM + (col & 63)] = (_Float16)val;
      } else {
        vT16[((size_t)(col >> 6) * HDIM + (col & 63)) * N_NODES + row] = (_Float16)val;
      }
    }
  }
}

// ---------------------------------------------------------------------------
// Stage 2: fused scores + softmax.  grid(256, 8), block 256, dynamic LDS.
// Block owns one (head, 16-row) strip; wave w computes cols [w*512, w*512+512)
// via WMMA (Q strip as A, K rows as column-major B), accumulates into LDS,
// then block-wide stable softmax, writing normalized P once to d_out.
// K-fragment loads double-buffered across the 32 column-tiles.
// ---------------------------------------------------------------------------
#define SC_LD 4104  // 4096 + 8 pad: breaks LDS bank alignment across rows

__global__ void attn_scores_softmax(const _Float16* __restrict__ q16,
                                    const _Float16* __restrict__ k16,
                                    float* __restrict__ attw) {
  extern __shared__ float smem[];
  float* sc     = smem;                 // [16][SC_LD]
  float* redmax = smem + 16 * SC_LD;    // [16][16]
  float* redsum = redmax + 256;         // [16][16]

  const int w    = threadIdx.x >> 5;
  const int lane = threadIdx.x & 31;
  const int hh   = lane >> 4, c16 = lane & 15;
  const int h    = blockIdx.y;
  const int n0   = blockIdx.x * 16;

  const _Float16* Qp = q16 + ((size_t)h * N_NODES + n0) * HDIM;
  const _Float16* Kp = k16 + (size_t)h * N_NODES * HDIM;

  // Q strip fragments for the whole d=64 reduction (2 k-steps), loaded once.
  const v16h a0 = load_a16(Qp, HDIM);
  const v16h a1 = load_a16(Qp + 32, HDIM);

  v16h b0 = load_bt16(Kp + (size_t)(w * 512) * HDIM, HDIM);
  v16h b1 = load_bt16(Kp + (size_t)(w * 512) * HDIM + 32, HDIM);

  for (int ct = 0; ct < 32; ++ct) {
    const int coln = w * 512 + ((ct + 1) & 31) * 16;  // wrap: always in-bounds
    v16h nb0 = load_bt16(Kp + (size_t)coln * HDIM, HDIM);
    v16h nb1 = load_bt16(Kp + (size_t)coln * HDIM + 32, HDIM);

    v8f acc = wmma_f16(a0, b0, vzero8());
    acc = wmma_f16(a1, b1, acc);

    const int col0 = w * 512 + ct * 16;
#pragma unroll
    for (int e = 0; e < 8; ++e)
      sc[(size_t)(e + 8 * hh) * SC_LD + col0 + c16] = acc[e];

    b0 = nb0;
    b1 = nb1;
  }
  __syncthreads();

  // Stable softmax: 16 threads per row.
  const int row = threadIdx.x >> 4;
  const int t16 = threadIdx.x & 15;
  float m = -3.0e38f;
  for (int j = t16; j < N_NODES; j += 16)
    m = fmaxf(m, sc[(size_t)row * SC_LD + j]);
  redmax[row * 16 + t16] = m;
  __syncthreads();
  if (t16 == 0) {
    float mm = redmax[row * 16];
    for (int i = 1; i < 16; ++i) mm = fmaxf(mm, redmax[row * 16 + i]);
    redmax[row * 16] = mm;
  }
  __syncthreads();
  const float rm = redmax[row * 16];
  float s = 0.0f;
  for (int j = t16; j < N_NODES; j += 16) {
    float e = __expf(sc[(size_t)row * SC_LD + j] - rm);
    sc[(size_t)row * SC_LD + j] = e;
    s += e;
  }
  redsum[row * 16 + t16] = s;
  __syncthreads();
  if (t16 == 0) {
    float ss = 0.0f;
    for (int i = 0; i < 16; ++i) ss += redsum[row * 16 + i];
    redsum[row * 16] = ss;
  }
  __syncthreads();
  const float inv = 1.0f / redsum[row * 16];
  float* dst = attw + ((size_t)h * N_NODES + n0 + row) * N_NODES;
  for (int j = t16; j < N_NODES; j += 16)
    dst[j] = sc[(size_t)row * SC_LD + j] * inv;
}

// ---------------------------------------------------------------------------
// Stage 3: attended = P @ V.  grid(32, 8), block 256 (8 waves).
// Wave w: rows bx*128 + w*16, all 64 head-cols (4 tiles). P (fp32, from
// d_out) converted to f16 A-fragments on the fly; V^T is the B operand.
// Double-buffered across the 128 k-steps; prefetch runs ~32 steps ahead.
// ---------------------------------------------------------------------------
__global__ void pv_gemm(const float* __restrict__ attw,
                        const _Float16* __restrict__ vT16,
                        _Float16* __restrict__ att16) {
  const int w    = threadIdx.x >> 5;
  const int lane = threadIdx.x & 31;
  const int hh   = lane >> 4, c16 = lane & 15;
  const int h    = blockIdx.y;
  const int n0   = blockIdx.x * 128 + w * 16;

  const float* P = attw + ((size_t)h * N_NODES + n0) * N_NODES;
  const _Float16* Vt = vT16 + (size_t)h * HDIM * N_NODES;

  v8f acc[4];
#pragma unroll
  for (int t = 0; t < 4; ++t) acc[t] = vzero8();

  v16h a = load_a32(P, N_NODES);
  v16h b[4];
#pragma unroll
  for (int t = 0; t < 4; ++t) b[t] = load_bt16(Vt + (size_t)(t * 16) * N_NODES, N_NODES);

  for (int kk = 0; kk < N_NODES / 32; ++kk) {
    const int kn = ((kk + 1) & (N_NODES / 32 - 1)) * 32;  // wrap: always in-bounds
    if (kk + 32 < N_NODES / 32)
      __builtin_prefetch(P + kk * 32 + 1024, 0, 0);  // global_prefetch_b8
    v16h an = load_a32(P + kn, N_NODES);
    v16h bn[4];
#pragma unroll
    for (int t = 0; t < 4; ++t) bn[t] = load_bt16(Vt + (size_t)(t * 16) * N_NODES + kn, N_NODES);
#pragma unroll
    for (int t = 0; t < 4; ++t) acc[t] = wmma_f16(a, b[t], acc[t]);
    a = an;
#pragma unroll
    for (int t = 0; t < 4; ++t) b[t] = bn[t];
  }

#pragma unroll
  for (int t = 0; t < 4; ++t) {
#pragma unroll
    for (int e = 0; e < 8; ++e) {
      const int row = n0 + e + 8 * hh;
      const int col = h * HDIM + t * 16 + c16;
      att16[(size_t)row * HID + col] = (_Float16)acc[t][e];
    }
  }
}

// ---------------------------------------------------------------------------
// Stage 4: output = attended @ Wo + bo.  grid(128), block 256.
// Wave w: row-tile (w>>2), col-tile (w&3): block = 32 rows x 64 cols.
// ---------------------------------------------------------------------------
__global__ void out_gemm(const _Float16* __restrict__ att16,
                         const _Float16* __restrict__ WoT,
                         const float* __restrict__ bo,
                         float* __restrict__ out) {
  const int w    = threadIdx.x >> 5;
  const int lane = threadIdx.x & 31;
  const int hh   = lane >> 4, c16 = lane & 15;
  const int n0   = blockIdx.x * 32 + (w >> 2) * 16;
  const int c0   = (w & 3) * 16;

  const _Float16* Ap = att16 + (size_t)n0 * HID;
  const _Float16* Bp = WoT + (size_t)c0 * HID;

  v8f acc = vzero8();
  v16h a = load_a16(Ap, HID);
  v16h b = load_bt16(Bp, HID);
  for (int kk = 0; kk < HID / 32; ++kk) {
    const int kn = ((kk + 1) & (HID / 32 - 1)) * 32;
    v16h an = load_a16(Ap + kn, HID);
    v16h bn = load_bt16(Bp + kn, HID);
    acc = wmma_f16(a, b, acc);
    a = an;
    b = bn;
  }
#pragma unroll
  for (int e = 0; e < 8; ++e) {
    const int row = n0 + e + 8 * hh;
    const int col = c0 + c16;
    out[(size_t)row * HDIM + col] = acc[e] + bo[col];
  }
}

// ---------------------------------------------------------------------------
// Workspace layout (bytes, all 256-aligned); total ~22.6 MB.
// ---------------------------------------------------------------------------
static constexpr size_t OFF_X16  = 0;                          // 4096*512*2
static constexpr size_t OFF_WQT  = OFF_X16  + 4194304;         // 512*512*2
static constexpr size_t OFF_WKT  = OFF_WQT  + 524288;
static constexpr size_t OFF_WVT  = OFF_WKT  + 524288;
static constexpr size_t OFF_WOT  = OFF_WVT  + 524288;          // 64*512*2
static constexpr size_t OFF_Q16  = OFF_WOT  + 65536;           // 8*4096*64*2
static constexpr size_t OFF_K16  = OFF_Q16  + 4194304;
static constexpr size_t OFF_VT16 = OFF_K16  + 4194304;
static constexpr size_t OFF_A16  = OFF_VT16 + 4194304;         // 4096*512*2

extern "C" void kernel_launch(void* const* d_in, const int* in_sizes, int n_in,
                              void* d_out, int out_size, void* d_ws, size_t ws_size,
                              hipStream_t stream) {
  const float* x  = (const float*)d_in[0];
  const float* Wq = (const float*)d_in[1];
  const float* bq = (const float*)d_in[2];
  const float* Wk = (const float*)d_in[3];
  const float* bk = (const float*)d_in[4];
  const float* Wv = (const float*)d_in[5];
  const float* bv = (const float*)d_in[6];
  const float* Wo = (const float*)d_in[7];
  const float* bo = (const float*)d_in[8];

  float* out  = (float*)d_out;                       // [4096*64]
  float* attw = out + (size_t)N_NODES * HDIM;        // [8*4096*4096]

  char* ws = (char*)d_ws;
  _Float16* x16   = (_Float16*)(ws + OFF_X16);
  _Float16* WqT   = (_Float16*)(ws + OFF_WQT);
  _Float16* WkT   = (_Float16*)(ws + OFF_WKT);
  _Float16* WvT   = (_Float16*)(ws + OFF_WVT);
  _Float16* WoT   = (_Float16*)(ws + OFF_WOT);
  _Float16* q16   = (_Float16*)(ws + OFF_Q16);
  _Float16* k16   = (_Float16*)(ws + OFF_K16);
  _Float16* vT16  = (_Float16*)(ws + OFF_VT16);
  _Float16* att16 = (_Float16*)(ws + OFF_A16);

  // Stage 0: casts
  cast_f32_to_f16<<<(N_NODES * HID + 255) / 256, 256, 0, stream>>>(x, x16, N_NODES * HID);
  cast_transpose_f16<<<(HID * HID + 255) / 256, 256, 0, stream>>>(Wq, WqT, HID, HID);
  cast_transpose_f16<<<(HID * HID + 255) / 256, 256, 0, stream>>>(Wk, WkT, HID, HID);
  cast_transpose_f16<<<(HID * HID + 255) / 256, 256, 0, stream>>>(Wv, WvT, HID, HID);
  cast_transpose_f16<<<(HID * HDIM + 255) / 256, 256, 0, stream>>>(Wo, WoT, HID, HDIM);

  // Stage 1: Q/K/V projections
  qkv_gemm<<<dim3(N_NODES / 128, HID / 64, 3), 256, 0, stream>>>(
      x16, WqT, WkT, WvT, bq, bk, bv, q16, k16, vT16);

  // Stage 2: fused scores + softmax (writes P once into d_out)
  const size_t smem_bytes = (size_t)(16 * SC_LD + 512) * sizeof(float);  // ~259 KB < 320 KB/WGP
  attn_scores_softmax<<<dim3(N_NODES / 16, NHEAD), 256, smem_bytes, stream>>>(q16, k16, attw);

  // Stage 3: attended = P @ V (reads P once)
  pv_gemm<<<dim3(N_NODES / 128, NHEAD), 256, 0, stream>>>(attw, vT16, att16);

  // Stage 4: output projection
  out_gemm<<<N_NODES / 32, 256, 0, stream>>>(att16, WoT, bo, out);
}
